// GraphUNet_47665547051551
// MI455X (gfx1250) — compile-verified, matrix operand order checked by reading, not run on hardware
//
#include <hip/hip_runtime.h>
#include <hip/hip_bf16.h>

// ---------------------------------------------------------------------------
// Graph U-Net forward for MI455X (gfx1250).
// GEMMs: pre-convert activations to bf16, pre-pack weights into per-lane WMMA
// fragment order, then fully-unrolled v_wmma_f32_16x16x32_bf16 with b128 loads.
// Edge aggregation = float4 gather + global atomic f32 scatter (L2-resident).
// Top-k pooling = on-device 4-pass radix select + atomic compaction.
// ---------------------------------------------------------------------------

typedef __attribute__((ext_vector_type(16))) __bf16 v16bf;
typedef __attribute__((ext_vector_type(8)))  __bf16 v8bf;
typedef __attribute__((ext_vector_type(8)))  float  v8f;

#define NN   100000L
#define EE   1600000L
#define CC   128L
#define EPSV 1e-5f

__device__ __forceinline__ unsigned short f2bfbits(float f) {
    union { float f; unsigned u; } c; c.f = f;
    return (unsigned short)((c.u + 0x7FFFu + ((c.u >> 16) & 1u)) >> 16); // RNE
}

// ---------------- activation f32 -> bf16 row-major ---------------------------
__global__ void cvt_bf16(const float* __restrict__ x, unsigned short* __restrict__ y,
                         long n) {
    long i = (long)blockIdx.x * blockDim.x + threadIdx.x;
    if (i < n) y[i] = f2bfbits(x[i]);
}

// ---------------- pack W[K x 128] into per-lane WMMA B-fragment order --------
// Layout: Wp[(((tileN*ksteps + ks)*32) + lane)*16 + j], j = fragment element.
__global__ void pack_W(const float* __restrict__ W, unsigned short* __restrict__ Wp,
                       int K) {
    const int ksteps = K >> 5;
    const int total  = 8 * ksteps * 512;
    int t = blockIdx.x * blockDim.x + threadIdx.x;
    if (t >= total) return;
    int j    = t & 15;
    int lane = (t >> 4) & 31;
    int rest = t >> 9;
    int ks   = rest % ksteps;
    int tn   = rest / ksteps;
    int col  = tn * 16 + (lane & 15);
    int kgrp = (lane >> 4) * 8;
    int v    = j >> 1;
    int k    = ks * 32 + kgrp + (v & 3) * 2 + ((v >> 2) * 16) + (j & 1);
    Wp[t] = f2bfbits(W[(long)k * 128 + col]);
}

// ---------------- WMMA GEMM: C[M x 128] = Abf[M x K] @ Wp -------------------
// 256 threads = 8 waves; wave w -> 16x16 tile at cols [16w,16w+16).
// K templated (128/256) so the K-loop fully unrolls into back-to-back WMMAs.
template <int K>
__global__ __launch_bounds__(256) void gemm_wmma(
        const unsigned short* __restrict__ Abf, const unsigned short* __restrict__ Wp,
        float* __restrict__ C, int M) {
    constexpr int KSTEPS = K >> 5;
    const int lane = threadIdx.x & 31;
    const int row  = blockIdx.x * 16 + (lane & 15);
    const int rowc = row < M ? row : M - 1;          // clamp: keep EXEC uniform
    const int kgrp = (lane >> 4) * 8;
    const v8bf* ap = (const v8bf*)(Abf + (long)rowc * K + kgrp);
    const unsigned short* wp =
        Wp + ((long)(threadIdx.x >> 5) * KSTEPS * 32 + lane) * 16;
    v8f acc = {};
#pragma unroll
    for (int ks = 0; ks < KSTEPS; ++ks) {
        v8bf a0 = ap[ks * 4];            // k = ks*32 + kgrp + [0,8)
        v8bf a1 = ap[ks * 4 + 2];        // k = ks*32 + kgrp + 16 + [0,8)
        v16bf a = __builtin_shufflevector(a0, a1, 0, 1, 2, 3, 4, 5, 6, 7,
                                          8, 9, 10, 11, 12, 13, 14, 15);
        v16bf b = *(const v16bf*)(wp + ks * 512);
        acc = __builtin_amdgcn_wmma_f32_16x16x32_bf16(false, a, false, b,
                                                      (short)0, acc, false, false);
    }
    const int col   = (threadIdx.x >> 5) * 16 + (lane & 15);
    const int rbase = blockIdx.x * 16 + (lane >> 4) * 8;
    float* cp = C + (long)rbase * 128 + col;
    if (blockIdx.x * 16 + 16 <= M) {     // wave-uniform: full tile fast path
#pragma unroll
        for (int v = 0; v < 8; ++v) cp[v * 128] = acc[v];
    } else {                             // last partial tile only
#pragma unroll
        for (int v = 0; v < 8; ++v)
            if (rbase + v < M) cp[v * 128] = acc[v];
    }
}

// ---------------- fills ------------------------------------------------------
__global__ void fill_f32_k(float* p, float v, long n) {
    long i = (long)blockIdx.x * blockDim.x + threadIdx.x;
    if (i < n) p[i] = v;
}
__global__ void fill_u32_k(unsigned* p, unsigned v, long n) {
    long i = (long)blockIdx.x * blockDim.x + threadIdx.x;
    if (i < n) p[i] = v;
}

// ---------------- degree / normalization ------------------------------------
__global__ void deg_kernel(const int* __restrict__ dst, const int* __restrict__ m,
                           float* __restrict__ deg, int E) {
    int e = blockIdx.x * blockDim.x + threadIdx.x;
    if (e >= E) return;
    if (m[e]) atomicAdd(&deg[dst[e]], 1.0f);
}
__global__ void dinv_kernel(const float* __restrict__ deg, float* __restrict__ dinv,
                            float* __restrict__ inv, int n) {
    int i = blockIdx.x * blockDim.x + threadIdx.x;
    if (i >= n) return;
    float d = deg[i];
    dinv[i] = rsqrtf(d);
    inv[i]  = 1.0f / d;
}

// ---------------- edge scatter-add: agg[dst] += h[src]*norm -----------------
__global__ void edge_agg(const float* __restrict__ h, const int* __restrict__ src,
                         const int* __restrict__ dst, const int* __restrict__ m,
                         const float* __restrict__ dinv, float* __restrict__ agg, int E) {
    long t = (long)blockIdx.x * blockDim.x + threadIdx.x;
    long e = t >> 5;
    int  c = (int)(t & 31) * 4;
    if (e >= E) return;
    if (!m[e]) return;
    int s = src[e], d = dst[e];
    float nrm = dinv[s] * dinv[d];
    const float4 hv = *(const float4*)(h + (long)s * 128 + c);
    float* ap = agg + (long)d * 128 + c;
    atomicAdd(ap + 0, hv.x * nrm);
    atomicAdd(ap + 1, hv.y * nrm);
    atomicAdd(ap + 2, hv.z * nrm);
    atomicAdd(ap + 3, hv.w * nrm);
}

// out = agg + h/deg + bias (optional relu)
__global__ void finalize_gcn(const float* __restrict__ agg, const float* __restrict__ h,
                             const float* __restrict__ inv, const float* __restrict__ bias,
                             float* __restrict__ out, int n, int relu) {
    long i = (long)blockIdx.x * blockDim.x + threadIdx.x;
    if (i >= (long)n * 128) return;
    int r = (int)(i >> 7), c = (int)(i & 127);
    float v = agg[i] + h[i] * inv[r] + bias[c];
    if (relu) v = fmaxf(v, 0.f);
    out[i] = v;
}

// ---------------- BatchNorm (training forward) + ReLU ------------------------
__global__ __launch_bounds__(256) void bn_stats(const float* __restrict__ x,
                                                float* __restrict__ mv, int n) {
    const int c = blockIdx.x;
    float s = 0.f, s2 = 0.f;
    for (int i = threadIdx.x; i < n; i += 256) {
        float v = x[(long)i * 128 + c];
        s += v; s2 += v * v;
    }
    __shared__ float sh[256], sh2[256];
    sh[threadIdx.x] = s; sh2[threadIdx.x] = s2;
    __syncthreads();
    for (int st = 128; st > 0; st >>= 1) {
        if (threadIdx.x < st) { sh[threadIdx.x] += sh[threadIdx.x + st];
                                sh2[threadIdx.x] += sh2[threadIdx.x + st]; }
        __syncthreads();
    }
    if (threadIdx.x == 0) {
        float mean = sh[0] / n;
        mv[c]       = mean;
        mv[128 + c] = sh2[0] / n - mean * mean;
    }
}
__global__ void bn_apply(float* __restrict__ x, const float* __restrict__ mv,
                         const float* __restrict__ g, const float* __restrict__ beta, int n) {
    long i = (long)blockIdx.x * blockDim.x + threadIdx.x;
    if (i >= (long)n * 128) return;
    int c = (int)(i & 127);
    float y = (x[i] - mv[c]) * rsqrtf(mv[128 + c] + EPSV) * g[c] + beta[c];
    x[i] = fmaxf(y, 0.f);
}

// ---------------- top-k pooling ----------------------------------------------
__global__ void wnorm_kernel(const float* __restrict__ w, float* __restrict__ wn) {
    __shared__ float sh[128];
    float v = w[threadIdx.x];
    sh[threadIdx.x] = v * v;
    __syncthreads();
    for (int s = 64; s > 0; s >>= 1) {
        if (threadIdx.x < s) sh[threadIdx.x] += sh[threadIdx.x + s];
        __syncthreads();
    }
    if (threadIdx.x == 0) *wn = sqrtf(sh[0]);
}
__global__ void score_kernel(const float* __restrict__ x, const float* __restrict__ w,
                             const float* __restrict__ wn, unsigned* __restrict__ keys,
                             float* __restrict__ score, int n) {
    int i = blockIdx.x * blockDim.x + threadIdx.x;
    if (i >= n) return;
    float d = 0.f;
#pragma unroll 8
    for (int c = 0; c < 128; ++c) d += x[(long)i * 128 + c] * w[c];
    float s = tanhf(d / *wn);
    score[i] = s;
    unsigned u = __float_as_uint(s);
    u = (u & 0x80000000u) ? ~u : (u | 0x80000000u);
    keys[i] = u;
}
__global__ void sel_init(unsigned* sel, unsigned k) {
    sel[0] = 0u; sel[1] = k; sel[2] = 0u; sel[3] = 0u;
}
__global__ void hist_kernel(const unsigned* __restrict__ keys, const unsigned* __restrict__ sel,
                            unsigned* __restrict__ hist, int n, int p) {
    int i = blockIdx.x * blockDim.x + threadIdx.x;
    if (i >= n) return;
    unsigned key = keys[i];
    unsigned hm  = (p == 3) ? 0u : (0xFFFFFFFFu << ((p + 1) * 8));
    if ((key & hm) == (sel[0] & hm))
        atomicAdd(&hist[(key >> (8 * p)) & 0xFFu], 1u);
}
__global__ void pick_kernel(const unsigned* __restrict__ hist, unsigned* sel, int p) {
    unsigned kr = sel[1], prefix = sel[0];
    for (int b = 255; b >= 0; --b) {
        unsigned h = hist[b];
        if (kr > h) kr -= h;
        else { prefix |= ((unsigned)b) << (8 * p); break; }
    }
    sel[0] = prefix; sel[1] = kr;
}
__global__ void mark_kernel(const unsigned* __restrict__ keys, unsigned* sel,
                            int* __restrict__ newid, int* __restrict__ kept,
                            int* __restrict__ idx, int n) {
    int i = blockIdx.x * blockDim.x + threadIdx.x;
    if (i >= n) return;
    unsigned key = keys[i], T = sel[0];
    int keep = 0;
    if (key > T) keep = 1;
    else if (key == T) { unsigned t = atomicAdd(&sel[3], 1u); if (t < sel[1]) keep = 1; }
    if (keep) {
        int pz = (int)atomicAdd(&sel[2], 1u);
        newid[i] = pz; kept[i] = 1; idx[pz] = i;
    } else { newid[i] = 0; kept[i] = 0; }
}
__global__ void gather_rows(const float* __restrict__ x, const int* __restrict__ idx,
                            float* __restrict__ x2, int k) {
    long t = (long)blockIdx.x * blockDim.x + threadIdx.x;
    if (t >= (long)k * 128) return;
    int j = (int)(t >> 7), c = (int)(t & 127);
    x2[t] = x[(long)idx[j] * 128 + c];
}
__global__ void relabel_kernel(const int* __restrict__ s, const int* __restrict__ d,
                               const int* __restrict__ m, const int* __restrict__ newid,
                               const int* __restrict__ kept, int* __restrict__ s2,
                               int* __restrict__ d2, int* __restrict__ m2, int E) {
    int e = blockIdx.x * blockDim.x + threadIdx.x;
    if (e >= E) return;
    int ss = s[e], dd = d[e];
    m2[e] = m[e] && kept[ss] && kept[dd];
    s2[e] = newid[ss];
    d2[e] = newid[dd];
}

// ---------------- unpool + concat --------------------------------------------
__global__ void scatter_unpool(const float* __restrict__ x, const int* __restrict__ idx,
                               float* __restrict__ xc, int k) {
    long t = (long)blockIdx.x * blockDim.x + threadIdx.x;
    if (t >= (long)k * 128) return;
    int j = (int)(t >> 7), c = (int)(t & 127);
    xc[(long)idx[j] * 256 + c] = x[t];
}
__global__ void copy_concat(const float* __restrict__ xs, float* __restrict__ xc, int pn) {
    long t = (long)blockIdx.x * blockDim.x + threadIdx.x;
    if (t >= (long)pn * 128) return;
    int i = (int)(t >> 7), c = (int)(t & 127);
    xc[(long)i * 256 + 128 + c] = xs[t];
}

// ---------------------------------------------------------------------------
static inline int cdiv(long a, long b) { return (int)((a + b - 1) / b); }

// scratch layout (4-byte elements)
#define O_XS0  0L
#define O_XS1  (O_XS0 + NN * CC)
#define O_XS2  (O_XS1 + 50000L * CC)
#define O_XA   (O_XS2 + 25000L * CC)
#define O_XB   (O_XA + NN * CC)
#define O_AGG  (O_XB + NN * CC)
#define O_XC   (O_AGG + NN * CC)
#define O_DEG  (O_XC + NN * 2 * CC)
#define O_DINV (O_DEG + NN)
#define O_INV  (O_DINV + NN)
#define O_SCR  (O_INV + NN)
#define O_KEY  (O_SCR + NN)
#define O_NID  (O_KEY + NN)
#define O_KPT  (O_NID + NN)
#define O_IDX1 (O_KPT + NN)
#define O_IDX2 (O_IDX1 + 50000L)
#define O_IDX3 (O_IDX2 + 25000L)
#define O_MSK0 (O_IDX3 + 12500L)
#define O_SRC1 (O_MSK0 + EE)
#define O_DST1 (O_SRC1 + EE)
#define O_MSK1 (O_DST1 + EE)
#define O_SRC2 (O_MSK1 + EE)
#define O_DST2 (O_SRC2 + EE)
#define O_MSK2 (O_DST2 + EE)
#define O_SRC3 (O_MSK2 + EE)
#define O_DST3 (O_SRC3 + EE)
#define O_MSK3 (O_DST3 + EE)
#define O_HIST (O_MSK3 + EE)
#define O_SEL  (O_HIST + 256L)
#define O_MV   (O_SEL + 16L)
#define O_WN   (O_MV + 256L)
#define O_ABF  (O_WN + 8L)           // NN*256 bf16 = NN*128 4B slots
#define O_WP   (O_ABF + NN * CC)     // packed weights (bf16): 6*16384 + 3*32768
#define WP128  16384L
#define WP256  32768L

extern "C" void kernel_launch(void* const* d_in, const int* in_sizes, int n_in,
                              void* d_out, int out_size, void* d_ws, size_t ws_size,
                              hipStream_t stream) {
    const float* x_in  = (const float*)d_in[0];
    const int*   ei    = (const int*)d_in[1];
    const float* in_W  = (const float*)d_in[2];
    const float* in_b  = (const float*)d_in[3];
    const float* dn_W  = (const float*)d_in[4];
    const float* dn_b  = (const float*)d_in[5];
    const float* dn_g  = (const float*)d_in[6];
    const float* dn_be = (const float*)d_in[7];
    const float* pw    = (const float*)d_in[8];
    const float* bot_W = (const float*)d_in[9];
    const float* bot_b = (const float*)d_in[10];
    const float* up_W  = (const float*)d_in[11];
    const float* up_b  = (const float*)d_in[12];
    const float* up_g  = (const float*)d_in[13];
    const float* up_be = (const float*)d_in[14];
    const float* out_W = (const float*)d_in[15];
    const float* out_b = (const float*)d_in[16];
    float* out = (float*)d_out;

    float* base = (float*)d_ws;
    float* XS0 = base + O_XS0;  float* XS1 = base + O_XS1;  float* XS2 = base + O_XS2;
    float* XA  = base + O_XA;   float* XB  = base + O_XB;   float* AGG = base + O_AGG;
    float* XC  = base + O_XC;
    float* DEG = base + O_DEG;  float* DINV = base + O_DINV; float* INVD = base + O_INV;
    float* SCR = base + O_SCR;
    unsigned* KEYS = (unsigned*)(base + O_KEY);
    int* NID  = (int*)(base + O_NID);   int* KPT = (int*)(base + O_KPT);
    int* IDX1 = (int*)(base + O_IDX1);  int* IDX2 = (int*)(base + O_IDX2);
    int* IDX3 = (int*)(base + O_IDX3);
    int* MSK0 = (int*)(base + O_MSK0);
    int* SRC1 = (int*)(base + O_SRC1);  int* DST1 = (int*)(base + O_DST1);
    int* MSK1 = (int*)(base + O_MSK1);
    int* SRC2 = (int*)(base + O_SRC2);  int* DST2 = (int*)(base + O_DST2);
    int* MSK2 = (int*)(base + O_MSK2);
    int* SRC3 = (int*)(base + O_SRC3);  int* DST3 = (int*)(base + O_DST3);
    int* MSK3 = (int*)(base + O_MSK3);
    unsigned* HIST = (unsigned*)(base + O_HIST);
    unsigned* SEL  = (unsigned*)(base + O_SEL);
    float* MV = base + O_MV;  float* WN = base + O_WN;
    unsigned short* ABF = (unsigned short*)(base + O_ABF);
    unsigned short* WP  = (unsigned short*)(base + O_WP);

    unsigned short* WPin  = WP;
    unsigned short* WPdn0 = WPin  + WP128;
    unsigned short* WPdn1 = WPdn0 + WP128;
    unsigned short* WPdn2 = WPdn1 + WP128;
    unsigned short* WPbot = WPdn2 + WP128;
    unsigned short* WPout = WPbot + WP128;
    unsigned short* WPup0 = WPout + WP128;
    unsigned short* WPup1 = WPup0 + WP256;
    unsigned short* WPup2 = WPup1 + WP256;

    const int* SRC0 = ei;
    const int* DST0 = ei + EE;
    const long nlev[4] = {100000L, 50000L, 25000L, 12500L};

    auto fillf = [&](float* p, float v, long cnt) {
        fill_f32_k<<<cdiv(cnt, 256), 256, 0, stream>>>(p, v, cnt);
    };
    auto packw = [&](const float* Wm, unsigned short* Wp, int K) {
        int total = 8 * (K >> 5) * 512;
        pack_W<<<cdiv(total, 256), 256, 0, stream>>>(Wm, Wp, K);
    };
    auto compute_deg = [&](const int* dstp, const int* mp, long nl) {
        fillf(DEG, 1.0f, nl);
        deg_kernel<<<cdiv(EE, 256), 256, 0, stream>>>(dstp, mp, DEG, (int)EE);
        dinv_kernel<<<cdiv(nl, 256), 256, 0, stream>>>(DEG, DINV, INVD, (int)nl);
    };
    auto gcn = [&](const float* X, const unsigned short* Wp, const float* bias,
                   const int* s, const int* d, const int* m,
                   long nl, int K, float* dest, int relu) {
        cvt_bf16<<<cdiv(nl * K, 256), 256, 0, stream>>>(X, ABF, nl * K);
        if (K == 128)
            gemm_wmma<128><<<cdiv(nl, 16), 256, 0, stream>>>(ABF, Wp, XB, (int)nl);
        else
            gemm_wmma<256><<<cdiv(nl, 16), 256, 0, stream>>>(ABF, Wp, XB, (int)nl);
        fillf(AGG, 0.f, nl * CC);
        edge_agg<<<cdiv(EE * 32, 256), 256, 0, stream>>>(XB, s, d, m, DINV, AGG, (int)EE);
        finalize_gcn<<<cdiv(nl * CC, 256), 256, 0, stream>>>(AGG, XB, INVD, bias, dest,
                                                             (int)nl, relu);
    };
    auto bn = [&](float* X, const float* g, const float* be, long nl) {
        bn_stats<<<128, 256, 0, stream>>>(X, MV, (int)nl);
        bn_apply<<<cdiv(nl * CC, 256), 256, 0, stream>>>(X, MV, g, be, (int)nl);
    };
    auto pool = [&](const float* X, const float* pwl, long nl, long k, int* idxp) {
        wnorm_kernel<<<1, 128, 0, stream>>>(pwl, WN);
        score_kernel<<<cdiv(nl, 256), 256, 0, stream>>>(X, pwl, WN, KEYS, SCR, (int)nl);
        sel_init<<<1, 1, 0, stream>>>(SEL, (unsigned)k);
        for (int p = 3; p >= 0; --p) {
            fill_u32_k<<<1, 256, 0, stream>>>(HIST, 0u, 256);
            hist_kernel<<<cdiv(nl, 256), 256, 0, stream>>>(KEYS, SEL, HIST, (int)nl, p);
            pick_kernel<<<1, 1, 0, stream>>>(HIST, SEL, p);
        }
        mark_kernel<<<cdiv(nl, 256), 256, 0, stream>>>(KEYS, SEL, NID, KPT, idxp, (int)nl);
        gather_rows<<<cdiv(k * CC, 256), 256, 0, stream>>>(X, idxp, XA, (int)k);
    };
    auto relabel = [&](const int* s, const int* d, const int* m,
                       int* s2, int* d2, int* m2) {
        relabel_kernel<<<cdiv(EE, 256), 256, 0, stream>>>(s, d, m, NID, KPT, s2, d2, m2,
                                                          (int)EE);
    };
    auto unpool = [&](const float* Xk, const int* idxp, const float* xs, long pn, long k) {
        fillf(XC, 0.f, pn * 2 * CC);
        scatter_unpool<<<cdiv(k * CC, 256), 256, 0, stream>>>(Xk, idxp, XC, (int)k);
        copy_concat<<<cdiv(pn * CC, 256), 256, 0, stream>>>(xs, XC, (int)pn);
    };

    // --------------- pack all weights once -----------------------------------
    packw(in_W, WPin, 128);
    packw(dn_W + 0 * 128 * 128, WPdn0, 128);
    packw(dn_W + 1 * 128 * 128, WPdn1, 128);
    packw(dn_W + 2 * 128 * 128, WPdn2, 128);
    packw(bot_W, WPbot, 128);
    packw(out_W, WPout, 128);
    packw(up_W + 0 * 256 * 128, WPup0, 256);
    packw(up_W + 1 * 256 * 128, WPup1, 256);
    packw(up_W + 2 * 256 * 128, WPup2, 256);

    // --------------- forward -------------------------------------------------
    fill_u32_k<<<cdiv(EE, 256), 256, 0, stream>>>((unsigned*)MSK0, 1u, EE);
    compute_deg(DST0, MSK0, nlev[0]);
    gcn(x_in, WPin, in_b, SRC0, DST0, MSK0, nlev[0], 128, XA, 0);

    // down 0
    gcn(XA, WPdn0, dn_b + 0, SRC0, DST0, MSK0, nlev[0], 128, XS0, 0);
    bn(XS0, dn_g + 0, dn_be + 0, nlev[0]);
    pool(XS0, pw + 0, nlev[0], nlev[1], IDX1);
    relabel(SRC0, DST0, MSK0, SRC1, DST1, MSK1);
    compute_deg(DST1, MSK1, nlev[1]);

    // down 1
    gcn(XA, WPdn1, dn_b + 128, SRC1, DST1, MSK1, nlev[1], 128, XS1, 0);
    bn(XS1, dn_g + 128, dn_be + 128, nlev[1]);
    pool(XS1, pw + 128, nlev[1], nlev[2], IDX2);
    relabel(SRC1, DST1, MSK1, SRC2, DST2, MSK2);
    compute_deg(DST2, MSK2, nlev[2]);

    // down 2
    gcn(XA, WPdn2, dn_b + 256, SRC2, DST2, MSK2, nlev[2], 128, XS2, 0);
    bn(XS2, dn_g + 256, dn_be + 256, nlev[2]);
    pool(XS2, pw + 256, nlev[2], nlev[3], IDX3);
    relabel(SRC2, DST2, MSK2, SRC3, DST3, MSK3);
    compute_deg(DST3, MSK3, nlev[3]);

    // bottleneck (relu)
    gcn(XA, WPbot, bot_b, SRC3, DST3, MSK3, nlev[3], 128, XA, 1);

    // up 0 -> level 2
    unpool(XA, IDX3, XS2, nlev[2], nlev[3]);
    compute_deg(DST2, MSK2, nlev[2]);
    gcn(XC, WPup0, up_b + 0, SRC2, DST2, MSK2, nlev[2], 256, XA, 0);
    bn(XA, up_g + 0, up_be + 0, nlev[2]);

    // up 1 -> level 1
    unpool(XA, IDX2, XS1, nlev[1], nlev[2]);
    compute_deg(DST1, MSK1, nlev[1]);
    gcn(XC, WPup1, up_b + 128, SRC1, DST1, MSK1, nlev[1], 256, XA, 0);
    bn(XA, up_g + 128, up_be + 128, nlev[1]);

    // up 2 -> level 0
    unpool(XA, IDX1, XS0, nlev[0], nlev[1]);
    compute_deg(DST0, MSK0, nlev[0]);
    gcn(XC, WPup2, up_b + 256, SRC0, DST0, MSK0, nlev[0], 256, XA, 0);
    bn(XA, up_g + 256, up_be + 256, nlev[0]);

    // output conv
    gcn(XA, WPout, out_b, SRC0, DST0, MSK0, nlev[0], 128, out, 0);
}
